// Discriminator_I2T_56599079027110
// MI455X (gfx1250) — matrix operand order repaired and checked by reading, not compiled, not run on hardware
//
#include <hip/hip_runtime.h>

// Problem dims (fixed by the reference)
#define B_   64
#define T_   512
#define E_   512
#define H_   1024
#define G3_  (3 * H_)        // 3072

typedef unsigned short ushort_t;
typedef __attribute__((ext_vector_type(16))) __bf16        v16bf;
typedef __attribute__((ext_vector_type(8)))  float         v8f;
typedef __attribute__((ext_vector_type(4)))  int           v4i;
typedef __attribute__((ext_vector_type(4)))  unsigned int  v4u;   // builtin 16B vector

// Address-space-qualified pointers for the async-to-LDS builtin
typedef __attribute__((address_space(1))) v4i* g_v4ip;   // global v4i*
typedef __attribute__((address_space(3))) v4i* l_v4ip;   // LDS v4i*

#if __has_builtin(__builtin_amdgcn_global_load_async_to_lds_b128) && \
    __has_builtin(__builtin_amdgcn_s_wait_asynccnt)
#define HAVE_ASYNC_LDS 1
#else
#define HAVE_ASYNC_LDS 0
#endif

// ---------- helpers ----------

__device__ __forceinline__ ushort_t f2bf(float f) {   // RNE
  unsigned u = __float_as_uint(f);
  return (ushort_t)((u + 0x7FFFu + ((u >> 16) & 1u)) >> 16);
}
__device__ __forceinline__ float bf2f(ushort_t u) {
  return __uint_as_float(((unsigned)u) << 16);
}

// A-fragment (16x32 bf16, MxK), contiguous k order: k = 16*(lane>>4) + s
__device__ __forceinline__ v16bf load_frag_a(const ushort_t* __restrict__ p,
                                             int ld, int row0, int k0, int lane) {
  const int r  = row0 + (lane & 15);
  const int kb = k0 + ((lane >> 4) << 4);
  const ushort_t* q = p + (long)r * ld + kb;
  union { v16bf v; v4u u[2]; } t;
  t.u[0] = *(const v4u*)(q);
  t.u[1] = *(const v4u*)(q + 8);
  return t.v;
}

// B-fragment (32x16 bf16, KxN) straight from global (used by k_xproj)
__device__ __forceinline__ v16bf load_frag_b(const ushort_t* __restrict__ p,
                                             int ld, int col0, int k0, int lane) {
  const int c  = col0 + (lane & 15);
  const int kb = k0 + ((lane >> 4) << 4);
  const ushort_t* q = p + (long)c * ld + kb;
  union { v16bf v; v4u u[2]; } t;
  t.u[0] = *(const v4u*)(q);
  t.u[1] = *(const v4u*)(q + 8);
  return t.v;
}

// B-fragment from LDS, fragment-packed layout: frag f, lane L at byte (f*32+L)*32.
// base derives from a __shared__ array; AS inference selects ds_load_b128.
__device__ __forceinline__ v16bf load_frag_lds(const ushort_t* base, int frag, int lane) {
  const ushort_t* q = base + ((unsigned)frag * 32u + (unsigned)lane) * 16u;
  union { v16bf v; v4u u[2]; } t;
  t.u[0] = *(const v4u*)(q);
  t.u[1] = *(const v4u*)(q + 8);
  return t.v;
}

__device__ __forceinline__ float sigmoidf_(float x) {
  return 1.0f / (1.0f + __expf(-x));
}

// ---------- kernels ----------

__global__ void k_init(float* h32a, float* h32b, ushort_t* hbfa, ushort_t* hbfb,
                       unsigned* cnt) {
  int i = blockIdx.x * blockDim.x + threadIdx.x;
  int stride = gridDim.x * blockDim.x;
  const int n = B_ * H_;
  for (int idx = i; idx < n; idx += stride) {
    h32a[idx] = 0.0f; h32b[idx] = 0.0f;
    hbfa[idx] = 0;    hbfb[idx] = 0;
  }
  if (i == 0) *cnt = 0u;
}

__global__ void k_f2bf(const float* __restrict__ in, ushort_t* __restrict__ out, long n) {
  long i = (long)blockIdx.x * blockDim.x + threadIdx.x;
  long stride = (long)gridDim.x * blockDim.x;
  for (; i < n; i += stride) out[i] = f2bf(in[i]);
}

// x_proj[b*T+t, g] = sum_e x[b,t,e]*W_ih[g,e] + b_ih[g], stored in bf16
__global__ void k_xproj(const ushort_t* __restrict__ xbf,   // [B*T, E] bf16
                        const ushort_t* __restrict__ wih,   // [3H, E] bf16
                        const float* __restrict__ bih,      // [3H]
                        ushort_t* __restrict__ xproj) {     // [B*T, 3H] bf16
  const int lane = threadIdx.x & 31;
  const int wave = threadIdx.x >> 5;              // 0..7
  const int row0 = blockIdx.y * 128 + wave * 16;
  const int n0   = blockIdx.x * 64;

  v8f c[4] = {};
  for (int k = 0; k < E_; k += 32) {
    v16bf a = load_frag_a(xbf, E_, row0, k, lane);
#pragma unroll
    for (int nt = 0; nt < 4; ++nt) {
      v16bf b = load_frag_b(wih, E_, n0 + nt * 16, k, lane);
      c[nt] = __builtin_amdgcn_wmma_f32_16x16x32_bf16(false, a, false, b,
                                                      (short)0, c[nt], false, false);
    }
  }

  const int lh = lane >> 4, ln = lane & 15;
#pragma unroll
  for (int nt = 0; nt < 4; ++nt) {
    const int col = n0 + nt * 16 + ln;
    const float bb = bih[col];
#pragma unroll
    for (int r = 0; r < 8; ++r) {
      const int row = row0 + r + 8 * lh;
      xproj[(long)row * G3_ + col] = f2bf(c[nt][r] + bb);
    }
  }
}

// Persistent recurrent kernel: 64 blocks x 128 threads (4 waves).
// Block's W_hh slice (3 gates x 16 rows x 1024 K bf16 = 96 KB) is staged ONCE
// into LDS in fragment-packed order (async global->LDS), then each of 512
// steps runs entirely out of LDS for the B operands.
__global__ void k_gru(const ushort_t* __restrict__ whh_bf,  // [3H, H] bf16
                      const float* __restrict__ bhh,        // [3H]
                      const ushort_t* __restrict__ xproj,   // [B*T, 3H] bf16
                      const int* __restrict__ seqlen,       // [B]
                      float* h32a, float* h32b,             // [B, H] f32 (dbl buf)
                      ushort_t* hbfa, ushort_t* hbfb,       // [B, H] bf16 (dbl buf)
                      unsigned* cnt) {
  extern __shared__ ushort_t ldsW[];                  // 3*32*32 frags * 32B = 96 KB

  const int lane = threadIdx.x & 31;
  const int wave = threadIdx.x >> 5;     // 0..3 -> M tile (batch rows)
  const int j0   = blockIdx.x * 16;      // hidden tile
  const int b0   = wave * 16;
  const int lh   = lane >> 4;
  const int ln   = lane & 15;
  const int j    = j0 + ln;

  // ---- one-time stage of W_hh slice into LDS (fragment-packed) ----
  // chunk c = (g*32 + kk)*32 + L  holds  W_hh[g*H + j0 + (L&15)][kk*32 + 16*(L>>4) .. +16)
  const int NCH = 3 * 32 * 32;
  for (int c = threadIdx.x; c < NCH; c += blockDim.x) {
    const int g  = c >> 10;
    const int rm = c & 1023;
    const int kk = rm >> 5;
    const int L  = rm & 31;
    const int col = j0 + (L & 15);
    const int kb  = kk * 32 + ((L >> 4) << 4);
    const ushort_t* src = whh_bf + (long)(g * H_ + col) * H_ + kb;
    ushort_t* dst = ldsW + (size_t)c * 16;
#if HAVE_ASYNC_LDS
    __builtin_amdgcn_global_load_async_to_lds_b128(
        (g_v4ip)(uintptr_t)src,
        (l_v4ip)(unsigned)(uintptr_t)dst, 0, 0);
    __builtin_amdgcn_global_load_async_to_lds_b128(
        (g_v4ip)(uintptr_t)(src + 8),
        (l_v4ip)(unsigned)(uintptr_t)(dst + 8), 0, 0);
#else
    *(v4u*)(dst)     = *(const v4u*)(src);
    *(v4u*)(dst + 8) = *(const v4u*)(src + 8);
#endif
  }
#if HAVE_ASYNC_LDS
  __builtin_amdgcn_s_wait_asynccnt(0);
#endif
  __syncthreads();

  const float bhr = bhh[0 * H_ + j];
  const float bhz = bhh[1 * H_ + j];
  const float bhn = bhh[2 * H_ + j];

  for (int t = 0; t < T_; ++t) {
    const float*    h32r = (t & 1) ? h32b : h32a;
    float*          h32w = (t & 1) ? h32a : h32b;
    const ushort_t* hbr  = (t & 1) ? hbfb : hbfa;
    ushort_t*       hbw  = (t & 1) ? hbfa : hbfb;

    v8f cr = {}, cz = {}, cn = {};
    v16bf a = load_frag_a(hbr, H_, b0, 0, lane);    // prefetch kk=0
#pragma unroll 4
    for (int kk = 0; kk < 32; ++kk) {
      v16bf acur = a;
      if (kk + 1 < 32) a = load_frag_a(hbr, H_, b0, (kk + 1) * 32, lane);
      v16bf br = load_frag_lds(ldsW, 0 * 32 + kk, lane);
      v16bf bz = load_frag_lds(ldsW, 1 * 32 + kk, lane);
      v16bf bn = load_frag_lds(ldsW, 2 * 32 + kk, lane);
      cr = __builtin_amdgcn_wmma_f32_16x16x32_bf16(false, acur, false, br, (short)0, cr, false, false);
      cz = __builtin_amdgcn_wmma_f32_16x16x32_bf16(false, acur, false, bz, (short)0, cz, false, false);
      cn = __builtin_amdgcn_wmma_f32_16x16x32_bf16(false, acur, false, bn, (short)0, cn, false, false);
    }

    // gate math: C/D element r is (M = r + 8*(lane>>4), N = lane&15)
#pragma unroll
    for (int r = 0; r < 8; ++r) {
      const int b = b0 + r + 8 * lh;
      const ushort_t* xp = xproj + ((long)(b * T_ + t)) * G3_;
      const float hr = cr[r] + bhr;
      const float hz = cz[r] + bhz;
      const float hn = cn[r] + bhn;
      const float rg = sigmoidf_(bf2f(xp[0 * H_ + j]) + hr);
      const float zg = sigmoidf_(bf2f(xp[1 * H_ + j]) + hz);
      const float ng = tanhf(bf2f(xp[2 * H_ + j]) + rg * hn);
      const float hp = h32r[b * H_ + j];
      const float hv = (t < seqlen[b]) ? ((1.0f - zg) * ng + zg * hp) : hp;
      h32w[b * H_ + j] = hv;
      hbw[b * H_ + j]  = f2bf(hv);
    }

    // device-wide step barrier (monotonic counter; 64 co-resident blocks)
    __threadfence();
    __syncthreads();
    if (threadIdx.x == 0) {
      __hip_atomic_fetch_add(cnt, 1u, __ATOMIC_RELEASE, __HIP_MEMORY_SCOPE_AGENT);
      const unsigned target = (unsigned)gridDim.x * (unsigned)(t + 1);
      while (__hip_atomic_load(cnt, __ATOMIC_ACQUIRE, __HIP_MEMORY_SCOPE_AGENT) < target)
        __builtin_amdgcn_s_sleep(2);
    }
    __syncthreads();
    __builtin_amdgcn_fence(__ATOMIC_ACQUIRE, "agent");
  }
}

// out[b] = sigmoid(h[b,:] . W_out + b_out)
__global__ void k_head(const float* __restrict__ h, const float* __restrict__ wout,
                       const float* __restrict__ bout, float* __restrict__ out) {
  const int b = blockIdx.x * blockDim.x + threadIdx.x;
  if (b >= B_) return;
  float acc = bout[0];
  for (int k = 0; k < H_; ++k) acc += h[b * H_ + k] * wout[k];
  out[b] = sigmoidf_(acc);
}

// ---------- launcher ----------

extern "C" void kernel_launch(void* const* d_in, const int* in_sizes, int n_in,
                              void* d_out, int out_size, void* d_ws, size_t ws_size,
                              hipStream_t stream) {
  const float* x      = (const float*)d_in[0];   // [B,T,E]
  const int*   seqlen = (const int*)  d_in[1];   // [B]
  const float* Wih    = (const float*)d_in[2];   // [3H,E]
  const float* Whh    = (const float*)d_in[3];   // [3H,H]
  const float* bih    = (const float*)d_in[4];   // [3H]
  const float* bhh    = (const float*)d_in[5];   // [3H]
  const float* Wout   = (const float*)d_in[6];   // [1,H]
  const float* bout   = (const float*)d_in[7];   // [1]
  float* out = (float*)d_out;                    // [B,1] f32

  char* ws = (char*)d_ws;
  size_t off = 0;
  auto alloc = [&](size_t bytes) -> void* {
    void* p = ws + off;
    off = (off + bytes + 255) & ~(size_t)255;
    return p;
  };

  unsigned* cnt   = (unsigned*)alloc(256);
  ushort_t* xbf   = (ushort_t*)alloc((size_t)B_ * T_ * E_ * 2);
  ushort_t* wihbf = (ushort_t*)alloc((size_t)G3_ * E_ * 2);
  ushort_t* whhbf = (ushort_t*)alloc((size_t)G3_ * H_ * 2);
  ushort_t* xproj = (ushort_t*)alloc((size_t)B_ * T_ * G3_ * 2);  // bf16
  float*    h32a  = (float*)   alloc((size_t)B_ * H_ * 4);
  float*    h32b  = (float*)   alloc((size_t)B_ * H_ * 4);
  ushort_t* hbfa  = (ushort_t*)alloc((size_t)B_ * H_ * 2);
  ushort_t* hbfb  = (ushort_t*)alloc((size_t)B_ * H_ * 2);

  k_init<<<64, 256, 0, stream>>>(h32a, h32b, hbfa, hbfb, cnt);

  {
    long nx = (long)B_ * T_ * E_;
    int gb = (int)((nx + 255) / 256); if (gb > 4096) gb = 4096;
    k_f2bf<<<gb, 256, 0, stream>>>(x, xbf, nx);
    long nw1 = (long)G3_ * E_;
    gb = (int)((nw1 + 255) / 256); if (gb > 4096) gb = 4096;
    k_f2bf<<<gb, 256, 0, stream>>>(Wih, wihbf, nw1);
    long nw2 = (long)G3_ * H_;
    gb = (int)((nw2 + 255) / 256); if (gb > 4096) gb = 4096;
    k_f2bf<<<gb, 256, 0, stream>>>(Whh, whhbf, nw2);
  }

  {
    dim3 grid(G3_ / 64, (B_ * T_) / 128);   // 48 x 256
    k_xproj<<<grid, 256, 0, stream>>>(xbf, wihbf, bih, xproj);
  }

  // persistent sequential GRU scan; 96 KB dynamic LDS for the W_hh slice
  k_gru<<<64, 128, 3 * 32 * 32 * 32, stream>>>(whhbf, bhh, xproj, seqlen,
                                               h32a, h32b, hbfa, hbfb, cnt);

  float* hfin = ((T_ & 1) == 0) ? h32a : h32b;
  k_head<<<1, 64, 0, stream>>>(hfin, Wout, bout, out);
}
